// TBStars2MoEDecoderLayer_45457933860925
// MI455X (gfx1250) — matrix operand order repaired and checked by reading, not compiled, not run on hardware
//
#include <hip/hip_runtime.h>
#include <hip/hip_bf16.h>
#include <stdint.h>

// ---------------------------------------------------------------------------
// CDNA5 (gfx1250) TBStars2 MoE decoder layer.
// All GEMMs use v_wmma_f32_16x16x32_bf16 (fp32 accum). LDS tiles are stored in
// fragment-major (swizzled) order so each WMMA operand load is ds_load_b128 x2
// per lane. Staging uses float4 global reads + v_cvt_pk_bf16_f32 pairs and
// vector LDS stores wherever the swizzle permits.
// ---------------------------------------------------------------------------

typedef __attribute__((ext_vector_type(16))) __bf16 v16bf;
typedef __attribute__((ext_vector_type(8)))  float  v8f;

namespace {
constexpr int kT   = 2048;   // tokens
constexpr int kH   = 2048;   // hidden
constexpr int kH3  = 6144;   // 3*H (qkv)
constexpr int kNH  = 16;     // heads
constexpr int kHD  = 128;    // head dim
constexpr int kE   = 16;     // experts
constexpr int kF   = 1408;   // expert ffn
constexpr int kF2  = 2816;   // 2*F (fused gate+up)
constexpr int kFS  = 5632;   // shared ffn
constexpr int kNS  = 2 * kT; // routed slots (top-2)
}

// ---------------------------------------------------------------------------
// Fragment-major LDS swizzles (ushort element offsets).
// A operand 16x32 (code's contiguous k order): lane = (m&15) + 16*(k>>4),
//                               elem j = k&15.
// B operand 32x16:              lane = (n&15) + 16*(k>>4), elem j = k&15.
// Tile region = [m/n tile of 16][lane 32][elem 16] = 512 ushorts per 16x32.
// ---------------------------------------------------------------------------
__device__ __forceinline__ int aswz(int m, int k) {
  return (m >> 4) * 512 + ((m & 15) + 16 * (k >> 4)) * 16 + (k & 15);
}
__device__ __forceinline__ int bswz(int n, int k) {
  return (n >> 4) * 512 + ((n & 15) + 16 * (k >> 4)) * 16 + (k & 15);
}

__device__ __forceinline__ v16bf frag_ld(const unsigned short* p) {
  return *(const v16bf*)p;  // 32B aligned by construction
}

// pack two fp32 -> one dword of two bf16 (one v_cvt_pk_bf16_f32)
__device__ __forceinline__ unsigned int pk2(float lo, float hi) {
  union { __bf16 b[2]; unsigned int u; } t;
  t.b[0] = (__bf16)lo;
  t.b[1] = (__bf16)hi;
  return t.u;
}

// convert 8 consecutive fp32 (32B aligned) -> bf16, one 16B LDS vector store
__device__ __forceinline__ void cvt_store8(unsigned short* dst,
                                           const float* __restrict__ src) {
  const float4 f0 = ((const float4*)src)[0];
  const float4 f1 = ((const float4*)src)[1];
  union { __bf16 b[8]; uint4 v; } t;
  t.b[0] = (__bf16)f0.x; t.b[1] = (__bf16)f0.y;
  t.b[2] = (__bf16)f0.z; t.b[3] = (__bf16)f0.w;
  t.b[4] = (__bf16)f1.x; t.b[5] = (__bf16)f1.y;
  t.b[6] = (__bf16)f1.z; t.b[7] = (__bf16)f1.w;
  *(uint4*)dst = t.v;
}
__device__ __forceinline__ void zero_store8(unsigned short* dst) {
  *(uint4*)dst = make_uint4(0u, 0u, 0u, 0u);
}

__device__ __forceinline__ v8f wmma_bf16(v16bf a, v16bf b, v8f c) {
  return __builtin_amdgcn_wmma_f32_16x16x32_bf16(false, a, false, b, (short)0, c,
                                                 false, false);
}

// ---------------------------------------------------------------------------
// Generic NN GEMM: C[M,N] = A[M,K] * B[K,N] (+ addend). 64x64 tile, 8 waves.
// ---------------------------------------------------------------------------
__global__ __launch_bounds__(256) void gemm_nn_kernel(
    const float* __restrict__ A, const float* __restrict__ B,
    const float* __restrict__ addend, float* __restrict__ C,
    int M, int N, int K) {
  __shared__ alignas(32) unsigned short As[2048];  // 64m x 32k, A-swizzled
  __shared__ alignas(32) unsigned short Bs[2048];  // 32k x 64n, B-swizzled
  const int tid = threadIdx.x, lane = tid & 31, wave = tid >> 5;
  const int m0 = blockIdx.y * 64, n0 = blockIdx.x * 64;
  const int wm = (wave >> 1) * 16, wn = (wave & 1) * 32;
  const int ar = tid >> 2, ak = (tid & 3) * 8;    // A: 64 rows x 4 chunks of 8
  const int kp2 = (tid >> 4) * 2;                 // B: k row-pair base (0..30)
  const int nq  = (tid & 15) * 4;                 // B: n col base
  v8f c0 = {}, c1 = {};
  for (int k0 = 0; k0 < K; k0 += 32) {
    {
      const int m = m0 + ar;
      unsigned short* dst = As + aswz(ar, ak);
      if (m < M) {
        const float* p = A + (size_t)m * K + k0 + ak;
        cvt_store8(dst, p);
        if (k0 + 32 < K) __builtin_prefetch(p + 32, 0, 1);
      } else {
        zero_store8(dst);
      }
    }
    {
      // KN-major weights: 2k x 4n micro-tile per thread. Consecutive k for a
      // fixed n are adjacent ushorts in the B-swizzle -> packed b32 stores.
      const float* p0 = B + (size_t)(k0 + kp2) * N + n0 + nq;
      const float* p1 = p0 + N;
      if (n0 + nq + 4 <= N) {
        const float4 r0 = *(const float4*)p0;
        const float4 r1 = *(const float4*)p1;
        const float a0[4] = {r0.x, r0.y, r0.z, r0.w};
        const float a1[4] = {r1.x, r1.y, r1.z, r1.w};
#pragma unroll
        for (int j = 0; j < 4; ++j)
          *(unsigned int*)&Bs[bswz(nq + j, kp2)] = pk2(a0[j], a1[j]);
      } else {
#pragma unroll
        for (int j = 0; j < 4; ++j) {
          const int n = n0 + nq + j;
          const float v0 = (n < N) ? p0[j] : 0.0f;
          const float v1 = (n < N) ? p1[j] : 0.0f;
          *(unsigned int*)&Bs[bswz(nq + j, kp2)] = pk2(v0, v1);
        }
      }
    }
    __syncthreads();
    v16bf a  = frag_ld(As + (wm >> 4) * 512 + lane * 16);
    v16bf b0 = frag_ld(Bs + (wn >> 4) * 512 + lane * 16);
    v16bf b1 = frag_ld(Bs + ((wn + 16) >> 4) * 512 + lane * 16);
    c0 = wmma_bf16(a, b0, c0);
    c1 = wmma_bf16(a, b1, c1);
    __syncthreads();
  }
  const int nc = lane & 15;
  const int mb = m0 + wm + 8 * (lane >> 4);
#pragma unroll
  for (int r = 0; r < 8; ++r) {
    const int m = mb + r;
    if (m >= M) continue;
    int n = n0 + wn + nc;
    if (n < N) {
      float v = c0[r];
      if (addend) v += addend[(size_t)m * N + n];
      C[(size_t)m * N + n] = v;
    }
    n = n0 + wn + 16 + nc;
    if (n < N) {
      float v = c1[r];
      if (addend) v += addend[(size_t)m * N + n];
      C[(size_t)m * N + n] = v;
    }
  }
}

// ---------------------------------------------------------------------------
// Flash attention: grid (T/64, NH). 64-query tile, 64-key tiles, WMMA QK^T and
// PV, online softmax, causal mask. K and V share one LDS buffer.
// ---------------------------------------------------------------------------
__global__ __launch_bounds__(256) void attn_kernel(const float* __restrict__ qkv,
                                                   float* __restrict__ out) {
  const int head = blockIdx.y;
  const int qt = blockIdx.x;
  const int qbase = qt * 64;
  __shared__ alignas(32) unsigned short Qs[8192];   // [kt4][mt4][lane][16] A-swz
  __shared__ alignas(32) unsigned short KVs[8192];  // K: [kt4][nt4][...] B-swz
                                                    // V: [kt2][nt8][...] B-swz
  __shared__ alignas(16) float Ss[64 * 64];
  __shared__ alignas(32) unsigned short Ps[4096];   // [kt2][mt4][lane][16] A-swz
  __shared__ float mrow[64], lrow[64], srow[64];
  const int tid = threadIdx.x, lane = tid & 31, wave = tid >> 5;
  const int rw = (wave >> 1) * 16;      // row-frag base (S and O)
  const int sn = (wave & 1) * 32;       // S col base (2 frags)
  const int cb = (wave & 1) * 64;       // O col base (4 frags)
  const float scale = 0.08838834764831845f;  // 1/sqrt(128)

  // Q tile: 1024 chunks of 8, A-swizzled, contiguous stores
  for (int cc = tid; cc < 1024; cc += 256) {
    const int r = cc >> 4, c8 = (cc & 15) * 8;
    cvt_store8(Qs + (c8 >> 5) * 2048 + aswz(r, c8 & 31),
               qkv + (size_t)(qbase + r) * kH3 + head * kHD + c8);
  }
  if (tid < 64) { mrow[tid] = -3.0e38f; lrow[tid] = 0.0f; }
  v8f oacc[4] = {};

  for (int kt = 0; kt <= qt; ++kt) {
    const int kbase = kt * 64;
    __syncthreads();  // prev PV / Q load done before overwriting KVs
    // K tile: B-swizzled (n = key, k = head dim), contiguous stores
    for (int cc = tid; cc < 1024; cc += 256) {
      const int r = cc >> 4, c8 = (cc & 15) * 8;
      cvt_store8(KVs + (c8 >> 5) * 2048 + bswz(r, c8 & 31),
                 qkv + (size_t)(kbase + r) * kH3 + kH + head * kHD + c8);
    }
    __syncthreads();
    // S = Q * K^T
    v8f s0 = {}, s1 = {};
    for (int d = 0; d < 128; d += 32) {
      const int kt4 = d >> 5;
      v16bf aq  = frag_ld(Qs + kt4 * 2048 + (rw >> 4) * 512 + lane * 16);
      v16bf bk0 = frag_ld(KVs + kt4 * 2048 + (sn >> 4) * 512 + lane * 16);
      v16bf bk1 = frag_ld(KVs + kt4 * 2048 + ((sn + 16) >> 4) * 512 + lane * 16);
      s0 = wmma_bf16(aq, bk0, s0);
      s1 = wmma_bf16(aq, bk1, s1);
    }
    {
      const int nc = lane & 15;
      const int mb = rw + 8 * (lane >> 4);
#pragma unroll
      for (int r = 0; r < 8; ++r) {
        Ss[(mb + r) * 64 + sn + nc] = s0[r] * scale;
        Ss[(mb + r) * 64 + sn + 16 + nc] = s1[r] * scale;
      }
    }
    __syncthreads();
    // V tile into KVs (B-swizzled: k = key, n = head dim). 2 keys x 4 dims per
    // chunk: key pairs are adjacent elements -> packed b32 stores.
    for (int cc = tid; cc < 1024; cc += 256) {
      const int r2 = (cc >> 5) * 2;        // key row pair
      const int cq = (cc & 31) * 4;        // head-dim base
      const float* p0 = qkv + (size_t)(kbase + r2) * kH3 + 2 * kH + head * kHD + cq;
      const float* p1 = p0 + kH3;
      const float4 f0 = *(const float4*)p0;
      const float4 f1 = *(const float4*)p1;
      const float a0[4] = {f0.x, f0.y, f0.z, f0.w};
      const float a1[4] = {f1.x, f1.y, f1.z, f1.w};
      const int hgrp = 16 * ((r2 & 31) >> 4);
      const int kbas = (r2 >> 5) * 4096;
#pragma unroll
      for (int j = 0; j < 4; ++j) {
        const int c = cq + j;
        const int off = kbas + (c >> 4) * 512 + ((c & 15) + hgrp) * 16 + (r2 & 15);
        *(unsigned int*)&KVs[off] = pk2(a0[j], a1[j]);
      }
    }
    if (tid < 64) {
      const int r = tid;
      const int qrow = qbase + r;
      int climit = qrow - kbase + 1;  // causal: keys kbase..qrow
      if (climit > 64) climit = 64;
      const float mold = mrow[r];
      const float4* s4p = (const float4*)(Ss + r * 64);
      float mt = -3.0e38f;
      for (int c4 = 0; c4 < 16; ++c4) {
        const float4 s4 = s4p[c4];
        const int c = c4 * 4;
        if (c + 0 < climit) mt = fmaxf(mt, s4.x);
        if (c + 1 < climit) mt = fmaxf(mt, s4.y);
        if (c + 2 < climit) mt = fmaxf(mt, s4.z);
        if (c + 3 < climit) mt = fmaxf(mt, s4.w);
      }
      const float mnew = fmaxf(mold, mt);
      const float sc = __expf(mold - mnew);
      float sum = 0.0f;
      // 8-aligned column chunks of one row are 16 contiguous bytes in the
      // A-swizzle -> one b128 store per chunk.
      for (int c8 = 0; c8 < 64; c8 += 8) {
        const float4 sa = s4p[c8 >> 2];
        const float4 sb = s4p[(c8 >> 2) + 1];
        const float sv[8] = {sa.x, sa.y, sa.z, sa.w, sb.x, sb.y, sb.z, sb.w};
        union { __bf16 b[8]; uint4 v; } t;
#pragma unroll
        for (int q = 0; q < 8; ++q) {
          const int c = c8 + q;
          const float pv = (c < climit) ? __expf(sv[q] - mnew) : 0.0f;
          t.b[q] = (__bf16)pv;
          sum += pv;
        }
        *(uint4*)&Ps[(c8 >> 5) * 2048 + aswz(r, c8 & 31)] = t.v;
      }
      lrow[r] = lrow[r] * sc + sum;
      mrow[r] = mnew;
      srow[r] = sc;
    }
    __syncthreads();
    // rescale running O, then O += P * V
    {
      const int mb = rw + 8 * (lane >> 4);
#pragma unroll
      for (int r = 0; r < 8; ++r) {
        const float f = srow[mb + r];
        oacc[0][r] *= f; oacc[1][r] *= f; oacc[2][r] *= f; oacc[3][r] *= f;
      }
    }
    for (int kk = 0; kk < 64; kk += 32) {
      v16bf ap = frag_ld(Ps + (kk >> 5) * 2048 + (rw >> 4) * 512 + lane * 16);
#pragma unroll
      for (int t2 = 0; t2 < 4; ++t2) {
        v16bf bv = frag_ld(KVs + (kk >> 5) * 4096 +
                           ((cb + t2 * 16) >> 4) * 512 + lane * 16);
        oacc[t2] = wmma_bf16(ap, bv, oacc[t2]);
      }
    }
  }
  // normalize and write out
  {
    const int nc = lane & 15;
    const int mb = rw + 8 * (lane >> 4);
#pragma unroll
    for (int r = 0; r < 8; ++r) {
      const float linv = 1.0f / lrow[mb + r];
#pragma unroll
      for (int t2 = 0; t2 < 4; ++t2) {
        out[(size_t)(qbase + mb + r) * kH + head * kHD + cb + t2 * 16 + nc] =
            oacc[t2][r] * linv;
      }
    }
  }
}

// ---------------------------------------------------------------------------
// MoE grouped GEMM 1: gu[slot, 2F] = xn[tok(slot), :] @ w1[e]^T  (NT, gather-A)
// ---------------------------------------------------------------------------
__global__ __launch_bounds__(256) void moe_gemm1_kernel(
    const float* __restrict__ xn, const float* __restrict__ w1,
    float* __restrict__ gu, const int* __restrict__ seg_off,
    const int* __restrict__ tok_of_slot) {
  const int e = blockIdx.z;
  const int seg0 = seg_off[e], seg1 = seg_off[e + 1];
  const int rows = seg1 - seg0;
  const int m0 = blockIdx.y * 64;
  if (m0 >= rows) return;
  const int n0 = blockIdx.x * 64;
  const float* W = w1 + (size_t)e * kF2 * kH;
  __shared__ alignas(32) unsigned short As[2048];
  __shared__ alignas(32) unsigned short Bs[2048];
  __shared__ int toks[64];
  const int tid = threadIdx.x, lane = tid & 31, wave = tid >> 5;
  const int wm = (wave >> 1) * 16, wn = (wave & 1) * 32;
  const int ar = tid >> 2, ak = (tid & 3) * 8;
  if (tid < 64) {
    const int s = seg0 + m0 + tid;
    toks[tid] = (s < seg1) ? tok_of_slot[s] : -1;
  }
  __syncthreads();
  const int tk = toks[ar];
  v8f c0 = {}, c1 = {};
  for (int k0 = 0; k0 < kH; k0 += 32) {
    {
      unsigned short* dst = As + aswz(ar, ak);
      if (tk >= 0) cvt_store8(dst, xn + (size_t)tk * kH + k0 + ak);
      else zero_store8(dst);
    }
    // NT weights: rows are contiguous in K -> contiguous swizzled stores
    cvt_store8(Bs + bswz(ar, ak), W + (size_t)(n0 + ar) * kH + k0 + ak);
    __syncthreads();
    v16bf a  = frag_ld(As + (wm >> 4) * 512 + lane * 16);
    v16bf b0 = frag_ld(Bs + (wn >> 4) * 512 + lane * 16);
    v16bf b1 = frag_ld(Bs + ((wn + 16) >> 4) * 512 + lane * 16);
    c0 = wmma_bf16(a, b0, c0);
    c1 = wmma_bf16(a, b1, c1);
    __syncthreads();
  }
  const int nc = lane & 15;
  const int mb = wm + 8 * (lane >> 4);
#pragma unroll
  for (int r = 0; r < 8; ++r) {
    const int m = m0 + mb + r;
    if (m < rows) {
      gu[(size_t)(seg0 + m) * kF2 + n0 + wn + nc] = c0[r];
      gu[(size_t)(seg0 + m) * kF2 + n0 + wn + 16 + nc] = c1[r];
    }
  }
}

// ---------------------------------------------------------------------------
// MoE grouped GEMM 2: moe_out[tok, :] += w(slot) * (act[slot,:] @ w2[e]^T)
// Exactly two commutative float-atomic adds per element -> deterministic.
// ---------------------------------------------------------------------------
__global__ __launch_bounds__(256) void moe_gemm2_kernel(
    const float* __restrict__ act, const float* __restrict__ w2,
    float* __restrict__ moe_out, const int* __restrict__ seg_off,
    const int* __restrict__ tok_of_slot, const float* __restrict__ w_of_slot) {
  const int e = blockIdx.z;
  const int seg0 = seg_off[e], seg1 = seg_off[e + 1];
  const int rows = seg1 - seg0;
  const int m0 = blockIdx.y * 64;
  if (m0 >= rows) return;
  const int n0 = blockIdx.x * 64;
  const float* W = w2 + (size_t)e * kH * kF;
  __shared__ alignas(32) unsigned short As[2048];
  __shared__ alignas(32) unsigned short Bs[2048];
  const int tid = threadIdx.x, lane = tid & 31, wave = tid >> 5;
  const int wm = (wave >> 1) * 16, wn = (wave & 1) * 32;
  const int ar = tid >> 2, ak = (tid & 3) * 8;
  v8f c0 = {}, c1 = {};
  for (int k0 = 0; k0 < kF; k0 += 32) {
    {
      unsigned short* dst = As + aswz(ar, ak);
      const int slot = seg0 + m0 + ar;
      if (slot < seg1) cvt_store8(dst, act + (size_t)slot * kF + k0 + ak);
      else zero_store8(dst);
    }
    cvt_store8(Bs + bswz(ar, ak), W + (size_t)(n0 + ar) * kF + k0 + ak);
    __syncthreads();
    v16bf a  = frag_ld(As + (wm >> 4) * 512 + lane * 16);
    v16bf b0 = frag_ld(Bs + (wn >> 4) * 512 + lane * 16);
    v16bf b1 = frag_ld(Bs + ((wn + 16) >> 4) * 512 + lane * 16);
    c0 = wmma_bf16(a, b0, c0);
    c1 = wmma_bf16(a, b1, c1);
    __syncthreads();
  }
  const int nc = lane & 15;
  const int mb = wm + 8 * (lane >> 4);
#pragma unroll
  for (int r = 0; r < 8; ++r) {
    const int m = m0 + mb + r;
    if (m < rows) {
      const int slot = seg0 + m;
      const int tk = tok_of_slot[slot];
      const float wgt = w_of_slot[slot];
      atomicAdd(&moe_out[(size_t)tk * kH + n0 + wn + nc], wgt * c0[r]);
      atomicAdd(&moe_out[(size_t)tk * kH + n0 + wn + 16 + nc], wgt * c1[r]);
    }
  }
}

// ---------------------------------------------------------------------------
// Elementwise / routing kernels
// ---------------------------------------------------------------------------
__global__ void rmsnorm_kernel(const float* __restrict__ x,
                               const float* __restrict__ g,
                               float* __restrict__ y) {
  const int row = blockIdx.x;
  const float* xr = x + (size_t)row * kH;
  float s = 0.0f;
  for (int i = threadIdx.x; i < kH; i += 256) { const float v = xr[i]; s += v * v; }
  __shared__ float red[256];
  red[threadIdx.x] = s;
  __syncthreads();
  for (int st = 128; st > 0; st >>= 1) {
    if ((int)threadIdx.x < st) red[threadIdx.x] += red[threadIdx.x + st];
    __syncthreads();
  }
  const float rs = rsqrtf(red[0] / (float)kH + 1e-6f);
  for (int i = threadIdx.x; i < kH; i += 256)
    y[(size_t)row * kH + i] = xr[i] * rs * g[i];
}

__global__ void router_topk_kernel(const float* __restrict__ logits,
                                   int* __restrict__ tidx,
                                   float* __restrict__ twgt,
                                   int* __restrict__ counts) {
  const int t = blockIdx.x * blockDim.x + threadIdx.x;
  if (t >= kT) return;
  float l[kE];
  float mx = -3.0e38f;
#pragma unroll
  for (int e = 0; e < kE; ++e) { l[e] = logits[(size_t)t * kE + e]; mx = fmaxf(mx, l[e]); }
#pragma unroll
  for (int e = 0; e < kE; ++e) l[e] = __expf(l[e] - mx);  // unnormalized probs
  int i0 = 0; float v0 = -1.0f;
#pragma unroll
  for (int e = 0; e < kE; ++e) if (l[e] > v0) { v0 = l[e]; i0 = e; }
  int i1 = (i0 == 0) ? 1 : 0; float v1 = -1.0f;
#pragma unroll
  for (int e = 0; e < kE; ++e) if (e != i0 && l[e] > v1) { v1 = l[e]; i1 = e; }
  const float inv = 1.0f / (v0 + v1);  // renormalized top-2 weights
  tidx[2 * t] = i0; tidx[2 * t + 1] = i1;
  twgt[2 * t] = v0 * inv; twgt[2 * t + 1] = v1 * inv;
  atomicAdd(&counts[i0], 1);
  atomicAdd(&counts[i1], 1);
}

__global__ void scan_kernel(const int* __restrict__ counts,
                            int* __restrict__ offs, int* __restrict__ fillpos) {
  if (threadIdx.x == 0) {
    int acc = 0;
    for (int e = 0; e < kE; ++e) { offs[e] = acc; fillpos[e] = acc; acc += counts[e]; }
    offs[kE] = acc;
  }
}

__global__ void scatter_kernel(const int* __restrict__ tidx,
                               const float* __restrict__ twgt,
                               int* __restrict__ fillpos,
                               int* __restrict__ tok, float* __restrict__ wslot) {
  const int i = blockIdx.x * blockDim.x + threadIdx.x;
  if (i >= kNS) return;
  const int e = tidx[i];
  const int pos = atomicAdd(&fillpos[e], 1);
  tok[pos] = i >> 1;
  wslot[pos] = twgt[i];
}

__global__ void moe_act_kernel(const float* __restrict__ gu, float* __restrict__ act) {
  const int i = blockIdx.x * blockDim.x + threadIdx.x;
  if (i >= kNS * kF) return;
  const int slot = i / kF, f = i % kF;
  const float g = gu[(size_t)slot * kF2 + f];
  const float u = gu[(size_t)slot * kF2 + kF + f];
  act[i] = (g / (1.0f + __expf(-g))) * u;  // silu(g) * u
}

__global__ void silu_mul_kernel(const float* __restrict__ g,
                                const float* __restrict__ u,
                                float* __restrict__ out, int n) {
  const int i = blockIdx.x * blockDim.x + threadIdx.x;
  if (i >= n) return;
  const float x = g[i];
  out[i] = (x / (1.0f + __expf(-x))) * u[i];
}

__global__ void add3_kernel(const float* __restrict__ a, const float* __restrict__ b,
                            const float* __restrict__ c, float* __restrict__ out, int n) {
  const int i = blockIdx.x * blockDim.x + threadIdx.x;
  if (i < n) out[i] = a[i] + b[i] + c[i];
}

__global__ void zero_f32_kernel(float* __restrict__ p, int n) {
  const int i = blockIdx.x * blockDim.x + threadIdx.x;
  if (i < n) p[i] = 0.0f;
}

__global__ void zero_i32_kernel(int* __restrict__ p, int n) {
  const int i = blockIdx.x * blockDim.x + threadIdx.x;
  if (i < n) p[i] = 0;
}

// ---------------------------------------------------------------------------
// Orchestration
// ---------------------------------------------------------------------------
extern "C" void kernel_launch(void* const* d_in, const int* in_sizes, int n_in,
                              void* d_out, int out_size, void* d_ws, size_t ws_size,
                              hipStream_t stream) {
  (void)in_sizes; (void)n_in; (void)out_size; (void)ws_size;
  const float* hidden = (const float*)d_in[0];
  const float* pre_g  = (const float*)d_in[1];
  const float* post_g = (const float*)d_in[2];
  const float* w_qkv  = (const float*)d_in[3];
  const float* w_o    = (const float*)d_in[4];
  const float* gate_w = (const float*)d_in[5];
  const float* w1     = (const float*)d_in[6];
  const float* w2     = (const float*)d_in[7];
  const float* ws_g   = (const float*)d_in[8];
  const float* ws_u   = (const float*)d_in[9];
  const float* ws_d   = (const float*)d_in[10];
  float* out = (float*)d_out;

  // workspace layout (floats)
  float* ws = (float*)d_ws;
  size_t off = 0;
  float* xn     = ws + off; off += (size_t)kT * kH;
  float* qkv    = ws + off; off += (size_t)kT * kH3;      // aliased as gu later
  float* attn   = ws + off; off += (size_t)kT * kH;
  float* h1     = ws + off; off += (size_t)kT * kH;
  float* logits = ws + off; off += (size_t)kT * kE;
  float* sg     = ws + off; off += (size_t)kT * kFS;
  float* su     = ws + off; off += (size_t)kT * kFS;
  float* sdown  = ws + off; off += (size_t)kT * kH;
  float* act    = ws + off; off += (size_t)kNS * kF;
  float* moe    = ws + off; off += (size_t)kT * kH;
  float* twgt   = ws + off; off += (size_t)kNS;
  float* wslot  = ws + off; off += (size_t)kNS;
  int* ip       = (int*)(ws + off);
  int* tidx     = ip;            ip += kNS;
  int* counts   = ip;            ip += kE;
  int* offs     = ip;            ip += kE + 1;
  int* fillpos  = ip;            ip += kE;
  int* tok      = ip;            ip += kNS;
  float* gu = qkv;  // qkv dead after attention; gu (NS x 2F) fits inside

  // 1) pre-attention RMSNorm
  rmsnorm_kernel<<<kT, 256, 0, stream>>>(hidden, pre_g, xn);
  // 2) fused qkv projection
  gemm_nn_kernel<<<dim3(kH3 / 64, kT / 64), 256, 0, stream>>>(
      xn, w_qkv, nullptr, qkv, kT, kH3, kH);
  // 3) flash attention (causal)
  attn_kernel<<<dim3(kT / 64, kNH), 256, 0, stream>>>(qkv, attn);
  // 4) output projection fused with residual add: h1 = hidden + attn @ w_o
  gemm_nn_kernel<<<dim3(kH / 64, kT / 64), 256, 0, stream>>>(
      attn, w_o, hidden, h1, kT, kH, kH);
  // 5) post-attention RMSNorm
  rmsnorm_kernel<<<kT, 256, 0, stream>>>(h1, post_g, xn);
  // 6) router logits
  gemm_nn_kernel<<<dim3(1, kT / 64), 256, 0, stream>>>(
      xn, gate_w, nullptr, logits, kT, kE, kH);
  // 7) routing: top-2 + renorm, per-expert slot lists
  zero_i32_kernel<<<1, 64, 0, stream>>>(counts, kE);
  zero_f32_kernel<<<(kT * kH + 255) / 256, 256, 0, stream>>>(moe, kT * kH);
  router_topk_kernel<<<(kT + 255) / 256, 256, 0, stream>>>(logits, tidx, twgt, counts);
  scan_kernel<<<1, 32, 0, stream>>>(counts, offs, fillpos);
  scatter_kernel<<<(kNS + 255) / 256, 256, 0, stream>>>(tidx, twgt, fillpos, tok, wslot);
  // 8) expert gate+up grouped GEMM (gather-A, NT weights)
  moe_gemm1_kernel<<<dim3(kF2 / 64, kNS / 64, kE), 256, 0, stream>>>(
      xn, w1, gu, offs, tok);
  // 9) silu(gate) * up
  moe_act_kernel<<<(kNS * kF + 255) / 256, 256, 0, stream>>>(gu, act);
  // 10) expert down grouped GEMM with weighted atomic scatter
  moe_gemm2_kernel<<<dim3(kH / 64, kNS / 64, kE), 256, 0, stream>>>(
      act, w2, moe, offs, tok, wslot);
  // 11) shared-expert MLP
  gemm_nn_kernel<<<dim3(kFS / 64, kT / 64), 256, 0, stream>>>(
      xn, ws_g, nullptr, sg, kT, kFS, kH);
  gemm_nn_kernel<<<dim3(kFS / 64, kT / 64), 256, 0, stream>>>(
      xn, ws_u, nullptr, su, kT, kFS, kH);
  silu_mul_kernel<<<(kT * kFS + 255) / 256, 256, 0, stream>>>(sg, su, sg, kT * kFS);
  gemm_nn_kernel<<<dim3(kH / 64, kT / 64), 256, 0, stream>>>(
      sg, ws_d, nullptr, sdown, kT, kH, kFS);
  // 12) out = h1 + experts_out + shared_out
  add3_kernel<<<(kT * kH + 255) / 256, 256, 0, stream>>>(h1, moe, sdown, out, kT * kH);
}
